// UnimolAttention_86182813761626
// MI455X (gfx1250) — matrix operand order repaired and checked by reading, not compiled, not run on hardware
//
#include <hip/hip_runtime.h>
#include <hip/hip_bf16.h>

// ---------------------------------------------------------------------------
// GQA attention for MI455X (gfx1250): all matmuls via v_wmma_f32_16x16x32_bf16,
// fp32 online softmax with DPP16 ROW_XMASK butterflies, bias fed as WMMA C
// operand, b128 strided staging, V pre-transposed by its projection GEMM.
// ---------------------------------------------------------------------------

typedef __attribute__((ext_vector_type(16))) __bf16 v16bf;
typedef __attribute__((ext_vector_type(8)))  __bf16 v8bf;
typedef __attribute__((ext_vector_type(8)))  float  v8f;

#define LOG2E 1.44269504088896340736f
#define SEQ   1024

// fp32 -> bf16 round-to-nearest-even
static __device__ __forceinline__ unsigned short f2bf(float f) {
    unsigned int u = __float_as_uint(f);
    u += 0x7FFFu + ((u >> 16) & 1u);
    return (unsigned short)(u >> 16);
}

// DPP16 xor-shuffle within a row of 16 lanes (ROW_XMASK = 0x160|mask).
// A 16-lane row is exactly one matrix row in the WMMA C-tile layout.
template<int MASK>
static __device__ __forceinline__ float dpp_xor_f(float x) {
    return __int_as_float(__builtin_amdgcn_update_dpp(
        0, __float_as_int(x), 0x160 | MASK, 0xf, 0xf, true));
}
static __device__ __forceinline__ float rowmax16(float x) {
    x = fmaxf(x, dpp_xor_f<1>(x));
    x = fmaxf(x, dpp_xor_f<2>(x));
    x = fmaxf(x, dpp_xor_f<4>(x));
    x = fmaxf(x, dpp_xor_f<8>(x));
    return x;
}
static __device__ __forceinline__ float rowsum16(float x) {
    x += dpp_xor_f<1>(x);
    x += dpp_xor_f<2>(x);
    x += dpp_xor_f<4>(x);
    x += dpp_xor_f<8>(x);
    return x;
}

// A-fragment (16x32 bf16), contiguous k order shared with B: lane row = lane&15,
// kbase = (lane<16)?0:16, elements {kbase..kbase+15} (contiguous)
static __device__ __forceinline__ v16bf load_afrag(const unsigned short* row_ptr, int kbase) {
    v8bf lo = *(const v8bf*)(row_ptr + kbase);
    v8bf hi = *(const v8bf*)(row_ptr + kbase + 8);
    return __builtin_shufflevector(lo, hi, 0,1,2,3,4,5,6,7,8,9,10,11,12,13,14,15);
}

// ---------------------------------------------------------------------------
// Generic tiled GEMM: C[M,N] = A[M,K] * B[K,N]
//   A: fp32 or bf16 (A_BF16). B: fp32 weights.
//   OUT_MODE: 0 = fp32 row-major, 1 = bf16 row-major,
//             2 = bf16 row-major pre-scaled by 0.125 (Q path),
//             3 = bf16 transposed per 1024-row batch: C[(row/1024)*N+col][row%1024]
//                 (V path -> Vt[(b,hkv,d)][key], frees attention of transposing)
// Block = 256 threads (8 waves), tile 128(M) x 64(N), k-step 32.
// ---------------------------------------------------------------------------
template<bool A_BF16, int OUT_MODE>
__global__ __launch_bounds__(256)
void gemm_bf16_wmma(const void* __restrict__ Av, const float* __restrict__ B,
                    void* __restrict__ Cv, int M, int N, int K) {
    __shared__ __align__(16) unsigned short As[128 * 32];   // [m][k]
    __shared__ __align__(16) unsigned short Bts[64 * 32];   // [n][k] (transposed)
    const int tid  = threadIdx.x;
    const int lane = tid & 31;
    const int wave = tid >> 5;
    const int half = lane >> 4;
    const int l16  = lane & 15;
    const int m0 = blockIdx.y * 128;
    const int n0 = blockIdx.x * 64;

    // strided staging coordinates (constant per thread -> immediate offsets)
    const int ra = tid >> 2,  ca = (tid & 3) * 8;     // A: 8 bf16 per thread per row-step 64
    const int kb = tid >> 4,  nb = (tid & 15) * 4;    // B: 4 fp32 per thread per k-step 16

    v8f acc[4] = {};
    for (int k0 = 0; k0 < K; k0 += 32) {
        // ---- stage A tile 128x32 (b128) ----
        if (A_BF16) {
            const unsigned short* A = (const unsigned short*)Av;
            #pragma unroll
            for (int j = 0; j < 2; ++j)
                *(uint4*)&As[(ra + 64 * j) * 32 + ca] =
                    *(const uint4*)&A[(size_t)(m0 + ra + 64 * j) * K + k0 + ca];
        } else {
            const float* A = (const float*)Av;
            #pragma unroll
            for (int j = 0; j < 2; ++j) {
                const float* g = &A[(size_t)(m0 + ra + 64 * j) * K + k0 + ca];
                float4 u = *(const float4*)g;
                float4 w = *(const float4*)(g + 4);
                uint4 p;
                p.x = (unsigned)f2bf(u.x) | ((unsigned)f2bf(u.y) << 16);
                p.y = (unsigned)f2bf(u.z) | ((unsigned)f2bf(u.w) << 16);
                p.z = (unsigned)f2bf(w.x) | ((unsigned)f2bf(w.y) << 16);
                p.w = (unsigned)f2bf(w.z) | ((unsigned)f2bf(w.w) << 16);
                *(uint4*)&As[(ra + 64 * j) * 32 + ca] = p;
            }
        }
        // ---- stage B tile 32x64 fp32, transposed into Bts[n][k] ----
        #pragma unroll
        for (int j = 0; j < 2; ++j) {
            float4 v = *(const float4*)&B[(size_t)(k0 + kb + 16 * j) * N + n0 + nb];
            Bts[(nb + 0) * 32 + kb + 16 * j] = f2bf(v.x);
            Bts[(nb + 1) * 32 + kb + 16 * j] = f2bf(v.y);
            Bts[(nb + 2) * 32 + kb + 16 * j] = f2bf(v.z);
            Bts[(nb + 3) * 32 + kb + 16 * j] = f2bf(v.w);
        }
        __syncthreads();

        const int akb = half ? 16 : 0;
        const int bkb = half ? 16 : 0;
        v16bf afrag = load_afrag(&As[(wave * 16 + l16) * 32], akb);
        #pragma unroll
        for (int t = 0; t < 4; ++t) {
            v16bf bfrag = *(const v16bf*)&Bts[(t * 16 + l16) * 32 + bkb];
            acc[t] = __builtin_amdgcn_wmma_f32_16x16x32_bf16(
                false, afrag, false, bfrag, (short)0, acc[t], false, false);
        }
        __syncthreads();
    }
    // ---- store C (C-tile layout: row = vgpr + 8*half, col = l16) ----
    #pragma unroll
    for (int t = 0; t < 4; ++t) {
        #pragma unroll
        for (int v = 0; v < 8; ++v) {
            int row = m0 + wave * 16 + v + half * 8;
            int col = n0 + t * 16 + l16;
            if (OUT_MODE == 0) {
                ((float*)Cv)[(size_t)row * N + col] = acc[t][v];
            } else {
                float x = (OUT_MODE == 2) ? acc[t][v] * 0.125f : acc[t][v];
                unsigned short hv = f2bf(x);
                if (OUT_MODE == 3)
                    ((unsigned short*)Cv)[(size_t)((row >> 10) * N + col) * 1024 + (row & 1023)] = hv;
                else
                    ((unsigned short*)Cv)[(size_t)row * N + col] = hv;
            }
        }
    }
}

// ---------------------------------------------------------------------------
// Flash attention: one wave per (b, h, 16-query tile). 4 waves / block,
// per-wave private LDS regions -> no block barriers needed (same-wave DS ops
// are in-order on CDNA5).
// ---------------------------------------------------------------------------
__global__ __launch_bounds__(128)
void attn_flash_wmma(const unsigned short* __restrict__ Q,   // [4096][1024] bf16 (pre-scaled 0.125)
                     const unsigned short* __restrict__ Kb,  // [4096][256]  bf16
                     const unsigned short* __restrict__ Vt,  // [4*256][1024] bf16 (d-major)
                     const float* __restrict__ bias,         // [4][16][1024][1024]
                     const int*   __restrict__ mask,         // [4][1][1024][1024]
                     unsigned short* __restrict__ O)         // [4096][1024] bf16
{
    __shared__ __align__(16) unsigned short smem[4 * 5632];
    const int lane = threadIdx.x & 31;
    const int wave = threadIdx.x >> 5;
    unsigned short* Qs  = smem + wave * 5632;   // [16][64]
    unsigned short* Ks  = Qs + 1024;            // [32][64]
    unsigned short* Vts = Qs + 3072;            // [64][32] (d-major V)
    unsigned short* Ps  = Qs + 5120;            // [16][32]

    const int wid  = blockIdx.x * 4 + wave;     // 0..4095
    const int qt   = wid & 63;
    const int h    = (wid >> 6) & 15;
    const int b    = wid >> 10;
    const int hkv  = h >> 2;                    // repeat_interleave: h // n_rep
    const int half = lane >> 4;
    const int l16  = lane & 15;
    const int akb  = half ? 16 : 0;
    const int bkb  = half ? 16 : 0;

    const int bs0 = b * SEQ + qt * 16;          // first query row in [4096]

    // ---- stage Q tile [16][64]: strided b128 copies ----
    {
        const int r = lane >> 3, c = (lane & 7) * 8;
        const unsigned short* g = &Q[(size_t)(bs0 + r) * 1024 + h * 64 + c];
        unsigned short* l = &Qs[r * 64 + c];
        #pragma unroll
        for (int j = 0; j < 4; ++j)
            *(uint4*)(l + j * 256) = *(const uint4*)(g + (size_t)j * 4096);
    }
    v16bf qf0 = load_afrag(&Qs[l16 * 64], akb);        // d = 0..31
    v16bf qf1 = load_afrag(&Qs[l16 * 64 + 32], akb);   // d = 32..63

    v8f o0 = {}, o1 = {}, o2 = {}, o3 = {};
    float mrun[8], lrun[8];
    #pragma unroll
    for (int v = 0; v < 8; ++v) { mrun[v] = -1e30f; lrun[v] = 0.f; }

    const float* biasp = bias + (((size_t)(b * 16 + h)) * SEQ + qt * 16) * SEQ;
    const int*   maskp = mask + ((size_t)b * SEQ + qt * 16) * SEQ;

    // per-lane staging coordinates (loop-invariant)
    const int rk = lane >> 3, ck = (lane & 7) * 8;   // K: [32][64]
    const int rv = lane >> 2, cv = (lane & 3) * 8;   // Vt: [64][32]
    const unsigned short* gkb = &Kb[(size_t)(b * SEQ + rk) * 256 + hkv * 64 + ck];
    const unsigned short* gvb = &Vt[(size_t)((b * 4 + hkv) * 64 + rv) * 1024 + cv];

    for (int kc = 0; kc < SEQ / 32; ++kc) {
        // ---- stage K chunk [32 keys][64 d] (b128) ----
        {
            const unsigned short* g = gkb + (size_t)kc * 32 * 256;
            unsigned short* l = &Ks[rk * 64 + ck];
            #pragma unroll
            for (int j = 0; j < 8; ++j)
                *(uint4*)(l + j * 256) = *(const uint4*)(g + (size_t)j * 1024);
        }
        // ---- stage V chunk [64 d][32 keys] straight from d-major Vt (b128) ----
        {
            const unsigned short* g = gvb + kc * 32;
            unsigned short* l = &Vts[rv * 32 + cv];
            #pragma unroll
            for (int j = 0; j < 8; ++j)
                *(uint4*)(l + j * 256) = *(const uint4*)(g + (size_t)j * 8192);
        }

        // ---- bias preload straight into the WMMA C operand ----
        const int key0 = kc * 32 + l16;
        const int key1 = key0 + 16;
        v8f sa, sb;
        #pragma unroll
        for (int v = 0; v < 8; ++v) {
            int row = v + half * 8;
            sa[v] = biasp[(size_t)row * SEQ + key0];
            sb[v] = biasp[(size_t)row * SEQ + key1];
        }
        if (kc + 1 < SEQ / 32)   // gfx1250 global_prefetch for next bias lines
            __builtin_prefetch(&biasp[(size_t)(half * 8) * SEQ + key0 + 32], 0, 1);

        // ---- scores = bias + Q K^T (Q pre-scaled by 1/sqrt(DH)) ----
        {
            v16bf b0 = *(const v16bf*)&Ks[l16 * 64 + bkb];
            sa = __builtin_amdgcn_wmma_f32_16x16x32_bf16(false, qf0, false, b0, (short)0, sa, false, false);
            v16bf b1 = *(const v16bf*)&Ks[l16 * 64 + 32 + bkb];
            sa = __builtin_amdgcn_wmma_f32_16x16x32_bf16(false, qf1, false, b1, (short)0, sa, false, false);
            v16bf b2 = *(const v16bf*)&Ks[(16 + l16) * 64 + bkb];
            sb = __builtin_amdgcn_wmma_f32_16x16x32_bf16(false, qf0, false, b2, (short)0, sb, false, false);
            v16bf b3 = *(const v16bf*)&Ks[(16 + l16) * 64 + 32 + bkb];
            sb = __builtin_amdgcn_wmma_f32_16x16x32_bf16(false, qf1, false, b3, (short)0, sb, false, false);
        }

        // ---- mask, then online softmax (DPP16 ROW_XMASK butterflies) ----
        float s0[8], s1[8];
        #pragma unroll
        for (int v = 0; v < 8; ++v) {
            int row = v + half * 8;
            s0[v] = maskp[(size_t)row * SEQ + key0] ? sa[v] : -1e9f;
            s1[v] = maskp[(size_t)row * SEQ + key1] ? sb[v] : -1e9f;
        }
        #pragma unroll
        for (int v = 0; v < 8; ++v) {
            float rm = rowmax16(fmaxf(s0[v], s1[v]));
            float mnew  = fmaxf(mrun[v], rm);
            float scale = exp2f((mrun[v] - mnew) * LOG2E);
            mrun[v] = mnew;
            float p0 = exp2f((s0[v] - mnew) * LOG2E);
            float p1 = exp2f((s1[v] - mnew) * LOG2E);
            float rs = rowsum16(p0 + p1);
            lrun[v] = lrun[v] * scale + rs;
            o0[v] *= scale; o1[v] *= scale; o2[v] *= scale; o3[v] *= scale;
            // re-layout exp(P) via LDS: C-tile layout -> A-fragment layout
            Ps[(v + half * 8) * 32 + l16]      = f2bf(p0);
            Ps[(v + half * 8) * 32 + 16 + l16] = f2bf(p1);
        }

        // ---- O += P V  (A = P 16x32, B = V 32x64 as 4 n-tiles) ----
        v16bf pf = load_afrag(&Ps[l16 * 32], akb);
        {
            v16bf vb0 = *(const v16bf*)&Vts[(0 * 16 + l16) * 32 + bkb];
            o0 = __builtin_amdgcn_wmma_f32_16x16x32_bf16(false, pf, false, vb0, (short)0, o0, false, false);
            v16bf vb1 = *(const v16bf*)&Vts[(1 * 16 + l16) * 32 + bkb];
            o1 = __builtin_amdgcn_wmma_f32_16x16x32_bf16(false, pf, false, vb1, (short)0, o1, false, false);
            v16bf vb2 = *(const v16bf*)&Vts[(2 * 16 + l16) * 32 + bkb];
            o2 = __builtin_amdgcn_wmma_f32_16x16x32_bf16(false, pf, false, vb2, (short)0, o2, false, false);
            v16bf vb3 = *(const v16bf*)&Vts[(3 * 16 + l16) * 32 + bkb];
            o3 = __builtin_amdgcn_wmma_f32_16x16x32_bf16(false, pf, false, vb3, (short)0, o3, false, false);
        }
    }

    // ---- normalize and store attn in [b, q, h, d] bf16 layout ----
    #pragma unroll
    for (int v = 0; v < 8; ++v) {
        float inv = 1.0f / lrun[v];
        int row = bs0 + v + half * 8;
        size_t base = (size_t)row * 1024 + h * 64;
        O[base + 0 * 16 + l16] = f2bf(o0[v] * inv);
        O[base + 1 * 16 + l16] = f2bf(o1[v] * inv);
        O[base + 2 * 16 + l16] = f2bf(o2[v] * inv);
        O[base + 3 * 16 + l16] = f2bf(o3[v] * inv);
    }
}

// ---------------------------------------------------------------------------
extern "C" void kernel_launch(void* const* d_in, const int* in_sizes, int n_in,
                              void* d_out, int out_size, void* d_ws, size_t ws_size,
                              hipStream_t stream) {
    const float* hidden = (const float*)d_in[0];
    const float* bias   = (const float*)d_in[1];
    const int*   mask   = (const int*)d_in[2];
    const float* Wq     = (const float*)d_in[3];
    const float* Wk     = (const float*)d_in[4];
    const float* Wv     = (const float*)d_in[5];
    const float* Wo     = (const float*)d_in[6];
    float* out = (float*)d_out;

    char* ws = (char*)d_ws;
    unsigned short* Qbf = (unsigned short*)(ws);                       // 8 MB
    unsigned short* Kbf = (unsigned short*)(ws + (8u << 20));          // 2 MB
    unsigned short* Vtb = (unsigned short*)(ws + (10u << 20));         // 2 MB
    unsigned short* Abf = (unsigned short*)(ws + (12u << 20));         // 8 MB

    dim3 blk(256);
    // QKV projections (fp32 in, bf16 out). Q pre-scaled, V transposed.
    gemm_bf16_wmma<false, 2><<<dim3(16, 32), blk, 0, stream>>>(hidden, Wq, Qbf, 4096, 1024, 1024);
    gemm_bf16_wmma<false, 1><<<dim3(4, 32),  blk, 0, stream>>>(hidden, Wk, Kbf, 4096, 256, 1024);
    gemm_bf16_wmma<false, 3><<<dim3(4, 32),  blk, 0, stream>>>(hidden, Wv, Vtb, 4096, 256, 1024);
    // flash attention: 4096 waves, 4 waves/block
    attn_flash_wmma<<<dim3(1024), dim3(128), 0, stream>>>(Qbf, Kbf, Vtb, bias, mask, Abf);
    // output projection (bf16 in, fp32 out)
    gemm_bf16_wmma<true, 0><<<dim3(16, 32), blk, 0, stream>>>(Abf, Wo, out, 4096, 1024, 1024);
}
